// ProteinAWDLSTM_91182155694925
// MI455X (gfx1250) — compile-verified, matrix operand order checked by reading, not run on hardware
//
#include <hip/hip_runtime.h>
#include <hip/hip_bf16.h>

// ---------------------------------------------------------------------------
// Types for WMMA (gfx1250): A/B = 16 bf16 per lane, C/D = 8 f32 per lane.
// ---------------------------------------------------------------------------
typedef __attribute__((ext_vector_type(16))) __bf16 bf16x16;
typedef __attribute__((ext_vector_type(8)))  float  floatx8;
typedef __attribute__((ext_vector_type(4)))  unsigned uint32x4;
typedef __attribute__((ext_vector_type(4)))  int      int32x4;
typedef __attribute__((ext_vector_type(8)))  int      int32x8;

union Frag {
    uint32x4 u[2];
    bf16x16  v;
};

#define TSTEPS 512
#define BATCH  128
#define EMB    1024
#define TCHUNK 64           // time steps per recurrence launch (G chunk)

// ---------------------------------------------------------------------------
// Utility kernels
// ---------------------------------------------------------------------------
__global__ void k_zero_u32(unsigned* p, int n) {
    for (int i = threadIdx.x; i < n; i += blockDim.x) p[i] = 0u;
}

__global__ void k_reset_mask(const int* __restrict__ ids,
                             unsigned char* __restrict__ rst, int total) {
    int gid = blockIdx.x * 256 + threadIdx.x;
    if (gid >= total) return;
    int t = gid >> 7;                 // / BATCH
    rst[gid] = (t >= 2 && ids[gid - BATCH] == 0) ? 1 : 0;
}

__global__ void k_f32_to_bf16(const float* __restrict__ src,
                              __bf16* __restrict__ dst, int n) {
    int gid = blockIdx.x * 256 + threadIdx.x;
    if (gid < n) dst[gid] = (__bf16)src[gid];
}

__global__ void k_bias_add(const float* __restrict__ bi,
                           const float* __restrict__ bh,
                           float* __restrict__ bias, int n) {
    int gid = blockIdx.x * 256 + threadIdx.x;
    if (gid < n) bias[gid] = bi[gid] + bh[gid];
}

// Pack W [N, K] (row-major f32) into bf16 B-fragment tiles for
// V_WMMA_F32_16X16X32_BF16 (ISA 7.12.2 layout).  B(k,n) = W[n,k].
__global__ void k_pack_w(const float* __restrict__ W,
                         __bf16* __restrict__ WP, int K, int total) {
    int gid = blockIdx.x * 256 + threadIdx.x;
    if (gid >= total) return;
    const int h    = gid & 15;
    const int lane = (gid >> 4) & 31;
    const int tile = gid >> 9;
    const int kTiles = K >> 5;
    const int tn = tile / kTiles;
    const int tk = tile - tn * kTiles;
    const int n  = tn * 16 + (lane & 15);
    const int sel = (lane & 16) ? 8 : 0;
    const int k  = tk * 32 + ((h < 8) ? (sel + h) : (16 + sel + (h - 8)));
    WP[gid] = (__bf16)W[(size_t)n * K + k];
}

// ---------------------------------------------------------------------------
// Time-parallel pre-activation GEMM: G = X @ W^T + (bi+bh), bf16 out.
// Block = 8 waves; wave computes a 16x64 tile; block computes 128x64.
// ---------------------------------------------------------------------------
__global__ __launch_bounds__(256) void k_gemm_pre(
    const __bf16* __restrict__ X, const __bf16* __restrict__ WP,
    const float* __restrict__ bias, __bf16* __restrict__ G, int K, int N)
{
    const int lane = threadIdx.x & 31;
    const int w    = threadIdx.x >> 5;
    const int m0   = blockIdx.x * 128 + w * 16;
    const int n0   = blockIdx.y * 64;
    const int row  = m0 + (lane & 15);
    const int selA = (lane & 16) ? 8 : 0;
    const int kTiles = K >> 5;
    const size_t tstride = (size_t)kTiles * 512;

    floatx8 acc[4];
    floatx8 zacc = {};
#pragma unroll
    for (int j = 0; j < 4; ++j) acc[j] = zacc;

    const __bf16* xrow = X + (size_t)row * K;
    const __bf16* wcol = WP + (size_t)(n0 >> 4) * tstride + lane * 16;

    for (int k0 = 0; k0 < K; k0 += 32) {
        Frag a;
        a.u[0] = *(const uint32x4*)(xrow + k0 + selA);
        a.u[1] = *(const uint32x4*)(xrow + k0 + 16 + selA);
        __builtin_prefetch((const void*)(xrow + k0 + 32), 0, 1);
        const __bf16* wb = wcol + (size_t)(k0 >> 5) * 512;
#pragma unroll
        for (int j = 0; j < 4; ++j) {
            Frag b;
            b.u[0] = *(const uint32x4*)(wb + j * tstride);
            b.u[1] = *(const uint32x4*)(wb + j * tstride + 8);
            acc[j] = __builtin_amdgcn_wmma_f32_16x16x32_bf16(
                false, a.v, false, b.v, (short)0, acc[j], false, false);
        }
    }

    const int hi = (lane >> 4) & 1;
    const int nn = lane & 15;
#pragma unroll
    for (int j = 0; j < 4; ++j) {
        const int ncol = n0 + j * 16 + nn;
        const float bv = bias[ncol];
#pragma unroll
        for (int r = 0; r < 8; ++r) {
            const int m = m0 + hi * 8 + r;
            G[(size_t)m * N + ncol] = (__bf16)(acc[j][r] + bv);
        }
    }
}

// ---------------------------------------------------------------------------
// Grid-wide barrier: one fresh counter slot per use (zeroed before launch).
// ---------------------------------------------------------------------------
__device__ __forceinline__ void grid_barrier(unsigned* ctr, unsigned nBlocks) {
    __syncthreads();
    __threadfence();
    if (threadIdx.x == 0) {
        atomicAdd(ctr, 1u);
        while (atomicAdd(ctr, 0u) < nBlocks) __builtin_amdgcn_s_sleep(2);
    }
    __syncthreads();
    __threadfence();
}

// ---------------------------------------------------------------------------
// Persistent recurrence over TCHUNK timesteps.
//   pre[b,n] = G[t,b,n] + (mask(h_{t-1}) @ Wh^T)[b,n]   then gate math.
// 64 blocks x 8 waves.  All 8 waves of a block share one 16-row M-tile, so
// the 16 x K h_{t-1} tile is staged once per step into LDS — via the Tensor
// Data Mover when available (tensor_load_to_lds + s_wait_tensorcnt), else a
// coalesced cooperative copy — and the hot loop reads A with ds_load_b128.
// ---------------------------------------------------------------------------
template <int TPW, int KK>
__global__ __launch_bounds__(256) void k_recurrence(
    const __bf16* __restrict__ G,    // [TCHUNK*BATCH, N]
    const __bf16* __restrict__ WhP,  // packed (N/16)*(KK/32) tiles
    __bf16* __restrict__ Y,          // [T*BATCH, KK] bf16 layer output
    float*  __restrict__ Yf,         // f32 output (layer 2) or nullptr
    float*  __restrict__ cbuf,       // [BATCH*KK]
    float*  __restrict__ pre,        // [BATCH*N]
    const unsigned char* __restrict__ rst,  // [T*BATCH]
    unsigned* __restrict__ bars, int t0)
{
    constexpr int N = 4 * KK;
    __shared__ alignas(16) __bf16 sA[16 * KK];   // 16 rows of h_{t-1}

    const int lane  = threadIdx.x & 31;
    const int w     = threadIdx.x >> 5;
    const int W     = blockIdx.x * 8 + w;
    const int m0    = (W >> 6) * 16;             // == (blockIdx.x/8)*16
    const int ntb   = (W & 63) * TPW;
    const int rowL  = lane & 15;                 // lane's row within M-tile
    const int selA  = (lane & 16) ? 8 : 0;
    constexpr int kTiles = KK >> 5;
    constexpr size_t tstride = (size_t)kTiles * 512;
    const int hi = (lane >> 4) & 1;
    const int nn = lane & 15;
    const unsigned nB = gridDim.x;
    const uint32x4 z4 = {0u, 0u, 0u, 0u};
    const floatx8 zacc = {};

    for (int tl = 0; tl < TCHUNK; ++tl) {
        const int t = t0 + tl;
        floatx8 acc[TPW];
#pragma unroll
        for (int j = 0; j < TPW; ++j) acc[j] = zacc;

        if (t > 0) {
            // ---- stage h_{t-1} rows [m0, m0+16) into LDS ----
#if __has_builtin(__builtin_amdgcn_tensor_load_to_lds)
            if (threadIdx.x < 32) {
                const unsigned long long ga = (unsigned long long)(const void*)
                    (Y + ((size_t)(t - 1) * BATCH + m0) * KK);
                uint32x4 g0;
                g0[0] = 1u;                                  // count=1, user D#
                g0[1] = (unsigned)(size_t)(void*)sA;         // lds_addr
                g0[2] = (unsigned)ga;                        // global_addr lo
                g0[3] = ((unsigned)(ga >> 32) & 0x01FFFFFFu) // addr[56:32]
                        | 0x80000000u;                       // type=2 (image)
                int32x8 g1;
                g1[0] = 1 << 16;                             // data_size=2B
                g1[1] = (int)((unsigned)KK << 16);           // tensor_dim0 lo16
                g1[2] = (int)((unsigned)(KK >> 16) | (16u << 16)); // dim0 hi | tensor_dim1=16
                g1[3] = (int)((unsigned)KK << 16);           // tile_dim0 = KK
                g1[4] = 16;                                  // tile_dim1 = 16
                g1[5] = KK;                                  // dim0_stride lo32
                g1[6] = 0;
                g1[7] = 0;
                int32x4 zg = {0, 0, 0, 0};
                int32x8 zg8 = {0, 0, 0, 0, 0, 0, 0, 0};
                __builtin_amdgcn_tensor_load_to_lds(g0, g1, zg, zg, zg8, 0);
                __builtin_amdgcn_s_wait_tensorcnt(0);
            }
            __syncthreads();
#else
            {
                const __bf16* hbase = Y + ((size_t)(t - 1) * BATCH + m0) * KK;
                constexpr int CHUNKS = 16 * (KK / 8);        // 16B chunks
                for (int c = threadIdx.x; c < CHUNKS; c += 256) {
                    const int r = c / (KK / 8);
                    const int g = c - r * (KK / 8);
                    *(uint32x4*)(sA + r * KK + g * 8) =
                        *(const uint32x4*)(hbase + (size_t)r * KK + g * 8);
                }
                __syncthreads();
            }
#endif
            const bool rz = rst[(size_t)t * BATCH + m0 + rowL] != 0;
            const __bf16* wcol = WhP + (size_t)ntb * tstride + lane * 16;
            const __bf16* arow = sA + rowL * KK;
            for (int k0 = 0; k0 < KK; k0 += 32) {
                Frag a;
                a.u[0] = *(const uint32x4*)(arow + k0 + selA);
                a.u[1] = *(const uint32x4*)(arow + k0 + 16 + selA);
                if (rz) { a.u[0] = z4; a.u[1] = z4; }
                const __bf16* wb = wcol + (size_t)(k0 >> 5) * 512;
#pragma unroll
                for (int j = 0; j < TPW; ++j) {
                    Frag b;
                    b.u[0] = *(const uint32x4*)(wb + j * tstride);
                    b.u[1] = *(const uint32x4*)(wb + j * tstride + 8);
                    acc[j] = __builtin_amdgcn_wmma_f32_16x16x32_bf16(
                        false, a.v, false, b.v, (short)0, acc[j], false, false);
                }
            }
        }

        // pre = acc + G[t]
        const __bf16* Grow = G + (size_t)tl * BATCH * N;
#pragma unroll
        for (int j = 0; j < TPW; ++j) {
            const int ncol = (ntb + j) * 16 + nn;
#pragma unroll
            for (int r = 0; r < 8; ++r) {
                const int m = m0 + hi * 8 + r;
                pre[(size_t)m * N + ncol] =
                    acc[j][r] + (float)Grow[(size_t)m * N + ncol];
            }
        }

        grid_barrier(bars + tl * 2, nB);

        // Gate math: i,f,o,g chunks (reference order), c masked by reset.
        constexpr int total = BATCH * KK;
        for (int u = blockIdx.x * 256 + threadIdx.x; u < total; u += 256 * (int)nB) {
            const int b = u / KK;
            const int j = u - b * KK;
            const float* pb = pre + (size_t)b * N;
            const float ig = 1.0f / (1.0f + __expf(-pb[j]));
            const float fg = 1.0f / (1.0f + __expf(-pb[KK + j]));
            const float og = 1.0f / (1.0f + __expf(-pb[2 * KK + j]));
            const float gg = tanhf(pb[3 * KK + j]);
            const float cp = (t == 0 || rst[(size_t)t * BATCH + b]) ? 0.0f : cbuf[u];
            const float cn = fg * cp + ig * gg;
            const float h  = og * tanhf(cn);
            cbuf[u] = cn;
            Y[((size_t)t * BATCH + b) * KK + j] = (__bf16)h;
            if (Yf) Yf[((size_t)t * BATCH + b) * KK + j] = h;
        }

        grid_barrier(bars + tl * 2 + 1, nB);
    }
}

// ---------------------------------------------------------------------------
// Host launcher
// ---------------------------------------------------------------------------
extern "C" void kernel_launch(void* const* d_in, const int* in_sizes, int n_in,
                              void* d_out, int out_size, void* d_ws, size_t ws_size,
                              hipStream_t stream) {
    (void)in_sizes; (void)n_in; (void)out_size; (void)ws_size;

    const float* inputs = (const float*)d_in[0];
    const int*   ids    = (const int*)d_in[1];

    const int din_arr[3]  = {EMB, 2048, 2048};
    const int dout_arr[3] = {2048, 2048, EMB};

    // Workspace carve-out
    char* ws = (char*)d_ws;
    size_t off = 0;
    auto carve = [&](size_t bytes) -> char* {
        char* p = ws + off;
        off = (off + bytes + 255) & ~(size_t)255;
        return p;
    };
    unsigned char* rst  = (unsigned char*)carve((size_t)TSTEPS * BATCH);
    float*    bias      = (float*)carve((size_t)4 * 2048 * 4);
    unsigned* bars      = (unsigned*)carve(4096);
    float*    cbuf      = (float*)carve((size_t)BATCH * 2048 * 4);
    float*    pre       = (float*)carve((size_t)BATCH * 8192 * 4);
    __bf16*   WiP       = (__bf16*)carve((size_t)8192 * 2048 * 2);
    __bf16*   WhP       = (__bf16*)carve((size_t)8192 * 2048 * 2);
    __bf16*   G         = (__bf16*)carve((size_t)TCHUNK * BATCH * 8192 * 2);
    __bf16*   bufA      = (__bf16*)carve((size_t)TSTEPS * BATCH * 2048 * 2);
    __bf16*   bufB      = (__bf16*)carve((size_t)TSTEPS * BATCH * 2048 * 2);

    const int totTB = TSTEPS * BATCH;
    k_reset_mask<<<(totTB + 255) / 256, 256, 0, stream>>>(ids, rst, totTB);

    const int nInp = totTB * EMB;
    k_f32_to_bf16<<<(nInp + 255) / 256, 256, 0, stream>>>(inputs, bufA, nInp);

    __bf16* X = bufA;
    for (int l = 0; l < 3; ++l) {
        const float* Wi = (const float*)d_in[2 + 4 * l + 0];
        const float* bi = (const float*)d_in[2 + 4 * l + 1];
        const float* Wh = (const float*)d_in[2 + 4 * l + 2];
        const float* bh = (const float*)d_in[2 + 4 * l + 3];
        const int din  = din_arr[l];
        const int dout = dout_arr[l];
        const int N    = 4 * dout;

        int tot = N * din;
        k_pack_w<<<(tot + 255) / 256, 256, 0, stream>>>(Wi, WiP, din, tot);
        tot = N * dout;
        k_pack_w<<<(tot + 255) / 256, 256, 0, stream>>>(Wh, WhP, dout, tot);
        k_bias_add<<<(N + 255) / 256, 256, 0, stream>>>(bi, bh, bias, N);

        __bf16* Yb = (X == bufA) ? bufB : bufA;
        float*  Yf = (l == 2) ? (float*)d_out : nullptr;

        const int nChunks = TSTEPS / TCHUNK;
        for (int c = 0; c < nChunks; ++c) {
            const __bf16* Xc = X + (size_t)c * TCHUNK * BATCH * din;
            dim3 gg(TCHUNK * BATCH / 128, N / 64);
            k_gemm_pre<<<gg, 256, 0, stream>>>(Xc, WiP, bias, G, din, N);
            k_zero_u32<<<1, 256, 0, stream>>>(bars, 1024);
            if (N == 8192)
                k_recurrence<8, 2048><<<64, 256, 0, stream>>>(
                    G, WhP, Yb, Yf, cbuf, pre, rst, bars, c * TCHUNK);
            else
                k_recurrence<4, 1024><<<64, 256, 0, stream>>>(
                    G, WhP, Yb, Yf, cbuf, pre, rst, bars, c * TCHUNK);
        }
        X = Yb;
    }
}
